// GraphNeuralNetwork_78314433675855
// MI455X (gfx1250) — matrix operand order repaired and checked by reading, not compiled, not run on hardware
//
#include <hip/hip_runtime.h>

typedef __attribute__((ext_vector_type(16))) _Float16 v16h;
typedef __attribute__((ext_vector_type(8)))  _Float16 v8h;
typedef __attribute__((ext_vector_type(8)))  float    v8f;

#define BN_EPS 1e-5f
#define KP 128   // padded K for all WMMA GEMMs (100->128, 128->128)
#define NKB (KP / 32)

// ---------------- utility ----------------
__global__ void fill_zero_f32(float* p, long n) {
    long i = (long)blockIdx.x * blockDim.x + threadIdx.x;
    long stride = (long)gridDim.x * blockDim.x;
    for (; i < n; i += stride) p[i] = 0.0f;
}

// ---------------- graph normalization ----------------
__global__ void deg_kernel(const int* __restrict__ row, int E, float* __restrict__ deg) {
    int i = blockIdx.x * blockDim.x + threadIdx.x;
    int stride = gridDim.x * blockDim.x;
    for (; i < E; i += stride) atomicAdd(&deg[row[i]], 1.0f);
}

__global__ void dinv_kernel(float* __restrict__ deg, int n) {
    int i = blockIdx.x * blockDim.x + threadIdx.x;
    if (i < n) {
        float d = deg[i];
        deg[i] = (d > 0.0f) ? rsqrtf(d) : 0.0f;
    }
}

__global__ void coef_kernel(const int* __restrict__ row, const int* __restrict__ col,
                            const float* __restrict__ ew, const float* __restrict__ dinv,
                            int E, float* __restrict__ coef) {
    int e = blockIdx.x * blockDim.x + threadIdx.x;
    if (e < E) coef[e] = dinv[row[e]] * ew[e] * dinv[col[e]];
}

// ---------------- SpMM: y[col] += coef * x[row], edge per wave ----------------
__global__ void spmm_kernel(const float* __restrict__ x, const int* __restrict__ row,
                            const int* __restrict__ col, const float* __restrict__ coef,
                            int E, int F, float* __restrict__ y) {
    int lane = threadIdx.x & 31;
    long w  = ((long)blockIdx.x * blockDim.x + threadIdx.x) >> 5;
    long nw = ((long)gridDim.x * blockDim.x) >> 5;
    for (long e = w; e < E; e += nw) {
        int r = row[e], c = col[e];
        float cf = coef[e];
        const float* xs = x + (long)r * F;
        float* yd = y + (long)c * F;
        for (int f = lane; f < F; f += 32)
            atomicAdd(&yd[f], cf * xs[f]);
    }
}

// ---------------- f32 -> f16 activation conversion (row-major, K padded) ------
__global__ void cvt_a_f16(const float* __restrict__ src, int n, int Fin,
                          _Float16* __restrict__ dst) {
    long idx = (long)blockIdx.x * blockDim.x + threadIdx.x;
    long tot = (long)n * KP;
    if (idx < tot) {
        int k = (int)(idx % KP);
        long i = idx / KP;
        dst[idx] = (k < Fin) ? (_Float16)src[i * Fin + k] : (_Float16)0.0f;
    }
}

// ---------------- W -> f16, packed in B-fragment order -----------------------
// Wpk layout: [jt][kbi][lane][q], q=0..15 contiguous halfs per lane (32B).
// k order used here (B 32x16 f16, same as A): lane<16 holds K=kb..kb+15 of column
// j0+lane; lane>=16 holds K=kb+16..kb+31 of column j0+lane-16.
__global__ void pack_w_f16(const float* __restrict__ W, int Fin, int Fo,
                           _Float16* __restrict__ Wpk) {
    int idx = blockIdx.x * blockDim.x + threadIdx.x;   // over Fo * KP
    if (idx < Fo * KP) {
        int q    = idx & 15;
        int lane = (idx >> 4) & 31;
        int t    = idx >> 9;
        int kbi  = t % NKB;
        int jt   = t / NKB;
        int k = kbi * 32 + ((lane >= 16) ? 16 : 0) + q;
        int j = jt * 16 + (lane & 15);
        Wpk[idx] = (k < Fin) ? (_Float16)W[k * Fo + j] : (_Float16)0.0f;
    }
}

// ---------------- WMMA GEMM: out[n,FO] = xh[n,KP] @ W + bias -----------------
// One wave per 16-row stripe; computes all FO/16 column tiles, so each A
// fragment feeds FO/16 wmma ops. Fully unrolled: NKB * FO/16 static wmma.
template <int FO>
__global__ void wmma_gemm_kernel(const _Float16* __restrict__ xh,
                                 const _Float16* __restrict__ Wpk,
                                 const float* __restrict__ bias,
                                 float* __restrict__ out) {
    constexpr int NJ = FO / 16;
    const int i0 = blockIdx.x * 16;
    const int lane = threadIdx.x;
    const int hl = lane & 15;
    const bool hi = lane >= 16;
    const int koffA = hi ? 16 : 0;   // A 16x32 f16: lane<16 -> K{0..15}, lane>=16 -> K{16..31}
    const long arow = (long)(i0 + hl) * KP;

    v8f acc[NJ] = {};

#pragma unroll
    for (int kbi = 0; kbi < NKB; ++kbi) {
        const int kb = kbi * 32;
        v8h alo = *(const v8h*)(xh + arow + kb + koffA);
        v8h ahi = *(const v8h*)(xh + arow + kb + koffA + 8);
        v16h a;
#pragma unroll
        for (int t = 0; t < 8; ++t) { a[t] = alo[t]; a[t + 8] = ahi[t]; }
#pragma unroll
        for (int jt = 0; jt < NJ; ++jt) {
            const v16h b = *(const v16h*)(Wpk + (((long)jt * NKB + kbi) * 32 + lane) * 16);
            acc[jt] = __builtin_amdgcn_wmma_f32_16x16x32_f16(
                          /*neg_a=*/false, a, /*neg_b=*/false, b,
                          /*c_mod=*/(short)0, acc[jt],
                          /*reuse_a=*/false, /*reuse_b=*/false);
        }
    }

    const int ibase = i0 + (hi ? 8 : 0);   // C/D: VGPR r -> M=r (lanes<16), M=r+8 (lanes>=16)
#pragma unroll
    for (int jt = 0; jt < NJ; ++jt) {
        const int j = jt * 16 + hl;
        const float bj = bias[j];
#pragma unroll
        for (int r = 0; r < 8; ++r)
            out[(long)(ibase + r) * FO + j] = acc[jt][r] + bj;
    }
}

// ---------------- batch norm ----------------
__global__ void colstats_kernel(const float* __restrict__ z, int n, int Fo,
                                float* __restrict__ sum, float* __restrict__ sumsq) {
    int j = threadIdx.x;          // blockDim.x == Fo
    int r0 = blockIdx.x * 128;
    float s = 0.0f, ss = 0.0f;
    for (int r = 0; r < 128; ++r) {
        int i = r0 + r;
        if (i < n) {
            float v = z[(long)i * Fo + j];
            s += v; ss += v * v;
        }
    }
    atomicAdd(&sum[j], s);
    atomicAdd(&sumsq[j], ss);
}

__global__ void bnparams_kernel(const float* __restrict__ sum, const float* __restrict__ sumsq,
                                const float* __restrict__ g, const float* __restrict__ be,
                                int n, int Fo, float* __restrict__ scale,
                                float* __restrict__ shift) {
    int j = threadIdx.x;
    if (j < Fo) {
        float inv_n = 1.0f / (float)n;
        float mu = sum[j] * inv_n;
        float var = sumsq[j] * inv_n - mu * mu;
        float rstd = rsqrtf(var + BN_EPS);
        float sc = g[j] * rstd;
        scale[j] = sc;
        shift[j] = be[j] - mu * sc;
    }
}

__global__ void bn_relu_kernel(float* __restrict__ z, const float* __restrict__ scale,
                               const float* __restrict__ shift, long tot, int Fo) {
    long idx = (long)blockIdx.x * blockDim.x + threadIdx.x;
    if (idx < tot) {
        int j = (int)(idx % Fo);
        float v = z[idx] * scale[j] + shift[j];
        z[idx] = v > 0.0f ? v : 0.0f;
    }
}

// ---------------- host ----------------
extern "C" void kernel_launch(void* const* d_in, const int* in_sizes, int n_in,
                              void* d_out, int out_size, void* d_ws, size_t ws_size,
                              hipStream_t stream) {
    const float* x0  = (const float*)d_in[0];
    const int*   ei  = (const int*)d_in[1];
    const float* ew  = (const float*)d_in[2];
    const float* W0  = (const float*)d_in[3];
    const float* b0  = (const float*)d_in[4];
    const float* g0  = (const float*)d_in[5];
    const float* be0 = (const float*)d_in[6];
    const float* W1  = (const float*)d_in[7];
    const float* b1  = (const float*)d_in[8];
    const float* g1  = (const float*)d_in[9];
    const float* be1 = (const float*)d_in[10];
    const float* W2  = (const float*)d_in[11];
    const float* b2  = (const float*)d_in[12];

    const int F0 = 100, H = 128, FO = 64;
    const int n = in_sizes[0] / F0;          // 50000 (multiple of 16)
    const int E = in_sizes[2];               // 800000
    const int* row = ei;
    const int* col = ei + E;

    // workspace carve
    char* wsp = (char*)d_ws;
    auto carve = [&](size_t bytes) -> void* {
        void* p = (void*)wsp;
        wsp += (bytes + 255) & ~(size_t)255;
        return p;
    };
    float*    dinv  = (float*)   carve((size_t)n * 4);
    float*    coef  = (float*)   carve((size_t)E * 4);
    float*    y     = (float*)   carve((size_t)n * H * 4);
    float*    z     = (float*)   carve((size_t)n * H * 4);
    _Float16* xh    = (_Float16*)carve((size_t)n * KP * 2);
    _Float16* Wpk   = (_Float16*)carve((size_t)KP * H * 2);
    float*    stats = (float*)   carve((size_t)4 * H * 4);
    float* sum   = stats;
    float* sumsq = stats + H;
    float* scale = stats + 2 * H;
    float* shift = stats + 3 * H;

    auto cdiv = [](long a, long b) { return (int)((a + b - 1) / b); };

    // degree^-1/2 and per-edge coefficients (shared across the 3 graph convs)
    fill_zero_f32<<<cdiv(n, 256), 256, 0, stream>>>(dinv, n);
    deg_kernel<<<cdiv(E, 256), 256, 0, stream>>>(row, E, dinv);
    dinv_kernel<<<cdiv(n, 256), 256, 0, stream>>>(dinv, n);
    coef_kernel<<<cdiv(E, 256), 256, 0, stream>>>(row, col, ew, dinv, E, coef);

    auto run_spmm = [&](const float* xin, int F, float* yout) {
        fill_zero_f32<<<cdiv((long)n * F, 256), 256, 0, stream>>>(yout, (long)n * F);
        spmm_kernel<<<4096, 256, 0, stream>>>(xin, row, col, coef, E, F, yout);
    };
    auto run_bn = [&](float* zbuf, const float* g, const float* be) {
        fill_zero_f32<<<1, 256, 0, stream>>>(sum, 2 * H);   // sum & sumsq contiguous
        colstats_kernel<<<cdiv(n, 128), H, 0, stream>>>(zbuf, n, H, sum, sumsq);
        bnparams_kernel<<<1, H, 0, stream>>>(sum, sumsq, g, be, n, H, scale, shift);
        bn_relu_kernel<<<cdiv((long)n * H, 256), 256, 0, stream>>>(zbuf, scale, shift,
                                                                   (long)n * H, H);
    };

    // ---- layer 0: conv -> linear(100->128) -> BN -> relu ----
    run_spmm(x0, F0, y);
    cvt_a_f16<<<cdiv((long)n * KP, 256), 256, 0, stream>>>(y, n, F0, xh);
    pack_w_f16<<<cdiv((long)KP * H, 256), 256, 0, stream>>>(W0, F0, H, Wpk);
    wmma_gemm_kernel<128><<<n / 16, 32, 0, stream>>>(xh, Wpk, b0, z);
    run_bn(z, g0, be0);

    // ---- layer 1: conv -> linear(128->128) -> BN -> relu ----
    run_spmm(z, H, y);
    cvt_a_f16<<<cdiv((long)n * KP, 256), 256, 0, stream>>>(y, n, H, xh);
    pack_w_f16<<<cdiv((long)KP * H, 256), 256, 0, stream>>>(W1, H, H, Wpk);
    wmma_gemm_kernel<128><<<n / 16, 32, 0, stream>>>(xh, Wpk, b1, z);
    run_bn(z, g1, be1);

    // ---- final layer: conv -> linear(128->64), straight to d_out ----
    run_spmm(z, H, y);
    cvt_a_f16<<<cdiv((long)n * KP, 256), 256, 0, stream>>>(y, n, H, xh);
    pack_w_f16<<<cdiv((long)KP * FO, 256), 256, 0, stream>>>(W2, H, FO, Wpk);
    wmma_gemm_kernel<64><<<n / 16, 32, 0, stream>>>(xh, Wpk, b2, (float*)d_out);

    (void)n_in; (void)out_size; (void)ws_size;
}
